// TransformerEncoder_14645838480107
// MI455X (gfx1250) — matrix operand order repaired and checked by reading, not compiled, not run on hardware
//
#include <hip/hip_runtime.h>

#define B_  8
#define S_  1024
#define D_  512
#define L_  6
#define H_  8
#define HD_ 64
#define M_  (B_ * S_)   // 8192 rows of activations

typedef __attribute__((ext_vector_type(16))) __bf16 v16bf;
typedef __attribute__((ext_vector_type(8)))  __bf16 v8bf;
typedef __attribute__((ext_vector_type(8)))  float  v8f;

// ---------------------------------------------------------------------------
// WMMA: D = A * B + C   (bf16 in, fp32 accumulate); all modifiers zero.
// ---------------------------------------------------------------------------
__device__ __forceinline__ v8f wmma_bf16(v16bf a, v16bf b, v8f c) {
  return __builtin_amdgcn_wmma_f32_16x16x32_bf16(false, a, false, b,
                                                 (short)0, c, false, false);
}

// ---------------------------------------------------------------------------
// bf16 fragment loaders (no conversion in the hot loop).
// A (16x32): lane=16*hi+lr holds row M=lr; elem e -> K=16*hi+e.
//            p = &X[row*ld + k0 + hi*16]
// B (32x16): lane=16*hi+lr holds col N=lr; elem e -> K=16*hi+e.
//            p = &Bcol[k0 + 16*hi]
// ---------------------------------------------------------------------------
__device__ __forceinline__ v16bf frag_a_bf(const __bf16* __restrict__ p) {
  v8bf a = *(const v8bf*)(p);
  v8bf b = *(const v8bf*)(p + 8);
  return __builtin_shufflevector(a, b, 0, 1, 2, 3, 4, 5, 6, 7, 8, 9, 10, 11,
                                 12, 13, 14, 15);
}

__device__ __forceinline__ v16bf frag_b_bf(const __bf16* __restrict__ p) {
  return *(const v16bf*)(p);
}

__device__ __forceinline__ v16bf frag_b_bf_strided(const __bf16* __restrict__ p,
                                                   int s) {
  v16bf f;
#pragma unroll
  for (int i = 0; i < 16; ++i) f[i] = p[i * (size_t)s];
  return f;
}

// ---------------------------------------------------------------------------
// One-shot converters (run once per launch, outside the hot loops).
// ---------------------------------------------------------------------------
__global__ __launch_bounds__(256) void cvt_kernel(const float* __restrict__ s,
                                                  __bf16* __restrict__ d) {
  const size_t i = ((size_t)blockIdx.x * 256 + threadIdx.x) * 8;
#pragma unroll
  for (int k = 0; k < 8; ++k) d[i + k] = (__bf16)s[i + k];
}

__global__ __launch_bounds__(256) void init_x_kernel(
    const float* __restrict__ xr_in, const float* __restrict__ xi_in,
    float* __restrict__ Xr, float* __restrict__ Xi, __bf16* __restrict__ Xbr,
    __bf16* __restrict__ Xbi) {
  const size_t i = (size_t)blockIdx.x * 256 + threadIdx.x;
  const float r = xr_in[i], m = xi_in[i];
  Xr[i] = r;
  Xi[i] = m;
  Xbr[i] = (__bf16)r;
  Xbi[i] = (__bf16)m;
}

// ---------------------------------------------------------------------------
// Complex linear:  Yr = Ar*Wr^T - Ai*Wi^T + br ; Yi = Ar*Wi^T + Ai*Wr^T + bi
// A, W in bf16; fp32 WMMA accumulators; subtraction done on accumulators in
// the epilogue (bf16 WMMA has no NEG modifiers).
// M=8192, N=K=512.  8 waves/WG, 32x32 per wave, 64x128 per WG.
// grid = (M/64, N/128), block = 256.
// mode: bit0 = write fp32 pair, bit1 = write bf16 pair, bit2 = CReLU.
// ---------------------------------------------------------------------------
__global__ __launch_bounds__(256) void clin_kernel(
    const __bf16* __restrict__ Ar, const __bf16* __restrict__ Ai,
    const __bf16* __restrict__ Wr, const __bf16* __restrict__ Wi,
    const float* __restrict__ br, const float* __restrict__ bi,
    float* __restrict__ Yrf, float* __restrict__ Yif,
    __bf16* __restrict__ Yrb, __bf16* __restrict__ Yib, int mode) {
  const int lane = threadIdx.x & 31;
  const int wave = threadIdx.x >> 5;
  const int wm = wave & 1;   // 2 waves across M
  const int wn = wave >> 1;  // 4 waves across N
  const int m0 = blockIdx.x * 64 + wm * 32;
  const int n0 = blockIdx.y * 128 + wn * 32;
  const int hi = lane >> 4;
  const int lr = lane & 15;

  v8f acc_rr[2][2], acc_ii[2][2], acc_i[2][2];
#pragma unroll
  for (int a = 0; a < 2; ++a)
#pragma unroll
    for (int b = 0; b < 2; ++b)
#pragma unroll
      for (int j = 0; j < 8; ++j) {
        acc_rr[a][b][j] = 0.f;
        acc_ii[a][b][j] = 0.f;
        acc_i[a][b][j] = 0.f;
      }

  for (int k0 = 0; k0 < D_; k0 += 32) {
    v16bf far[2], fai[2], fwr[2], fwi[2];
#pragma unroll
    for (int t = 0; t < 2; ++t) {
      const size_t arow = (size_t)(m0 + 16 * t + lr) * D_ + k0 + hi * 16;
      far[t] = frag_a_bf(Ar + arow);
      fai[t] = frag_a_bf(Ai + arow);
      const size_t brow = (size_t)(n0 + 16 * t + lr) * D_ + k0 + hi * 16;
      fwr[t] = frag_b_bf(Wr + brow);
      fwi[t] = frag_b_bf(Wi + brow);
    }
#pragma unroll
    for (int mi = 0; mi < 2; ++mi)
#pragma unroll
      for (int ni = 0; ni < 2; ++ni) {
        acc_rr[mi][ni] = wmma_bf16(far[mi], fwr[ni], acc_rr[mi][ni]);
        acc_ii[mi][ni] = wmma_bf16(fai[mi], fwi[ni], acc_ii[mi][ni]);
        acc_i[mi][ni] = wmma_bf16(far[mi], fwi[ni], acc_i[mi][ni]);
        acc_i[mi][ni] = wmma_bf16(fai[mi], fwr[ni], acc_i[mi][ni]);
      }
  }

#pragma unroll
  for (int mi = 0; mi < 2; ++mi)
#pragma unroll
    for (int ni = 0; ni < 2; ++ni) {
      const int col = n0 + 16 * ni + lr;
      const float bR = br[col], bI = bi[col];
#pragma unroll
      for (int j = 0; j < 8; ++j) {
        const int row = m0 + 16 * mi + j + 8 * hi;
        float vr = acc_rr[mi][ni][j] - acc_ii[mi][ni][j] + bR;
        float vi = acc_i[mi][ni][j] + bI;
        if (mode & 4) {
          vr = fmaxf(vr, 0.f);
          vi = fmaxf(vi, 0.f);
        }
        const size_t o = (size_t)row * D_ + col;
        if (mode & 1) {
          Yrf[o] = vr;
          Yif[o] = vi;
        }
        if (mode & 2) {
          Yrb[o] = (__bf16)vr;
          Yib[o] = (__bf16)vi;
        }
      }
    }
}

// ---------------------------------------------------------------------------
// Flash attention, softmax over Re(q k^H) = qr.kr + qi.ki; all q/k/v/out bf16.
// One WG (4 waves) per (b, h, 64-query block); each wave owns 16 query rows.
// grid = B*H*(S/64) = 1024, block = 128.
// ---------------------------------------------------------------------------
__global__ __launch_bounds__(128) void attn_kernel(
    const __bf16* __restrict__ qr, const __bf16* __restrict__ qi,
    const __bf16* __restrict__ kr, const __bf16* __restrict__ ki,
    const __bf16* __restrict__ vr, const __bf16* __restrict__ vi,
    __bf16* __restrict__ Or, __bf16* __restrict__ Oi) {
  const int lane = threadIdx.x & 31;
  const int wave = threadIdx.x >> 5;
  const int hi = lane >> 4;
  const int lr = lane & 15;

  const int blk = blockIdx.x;
  const int qb = blk & 15;       // S/64 query blocks
  const int h = (blk >> 4) & 7;  // head
  const int b = blk >> 7;        // batch
  const float scale = 0.125f;    // HD^-0.5

  const size_t base = (size_t)b * S_ * D_ + (size_t)h * HD_;
  const int q0 = qb * 64 + wave * 16;

  v16bf fqr[2], fqi[2];
#pragma unroll
  for (int t = 0; t < 2; ++t) {
    const size_t off = base + (size_t)(q0 + lr) * D_ + t * 32 + hi * 16;
    fqr[t] = frag_a_bf(qr + off);
    fqi[t] = frag_a_bf(qi + off);
  }

  v8f o_r[4], o_i[4];
#pragma unroll
  for (int d = 0; d < 4; ++d)
#pragma unroll
    for (int j = 0; j < 8; ++j) {
      o_r[d][j] = 0.f;
      o_i[d][j] = 0.f;
    }
  float mrow[8], lrow[8];
#pragma unroll
  for (int j = 0; j < 8; ++j) {
    mrow[j] = -1e30f;
    lrow[j] = 0.f;
  }

  __shared__ __bf16 Pl[4][16 * 64];  // per-wave P staging tile (bf16)
  __bf16* myP = Pl[wave];

  for (int n0 = 0; n0 < S_; n0 += 64) {
    // ---- logits 16x64 ----
    v8f lg[4];
#pragma unroll
    for (int ni = 0; ni < 4; ++ni) {
#pragma unroll
      for (int j = 0; j < 8; ++j) lg[ni][j] = 0.f;
#pragma unroll
      for (int t = 0; t < 2; ++t) {
        const size_t boff =
            base + (size_t)(n0 + 16 * ni + lr) * D_ + t * 32 + hi * 16;
        lg[ni] = wmma_bf16(fqr[t], frag_b_bf(kr + boff), lg[ni]);
        lg[ni] = wmma_bf16(fqi[t], frag_b_bf(ki + boff), lg[ni]);
      }
    }
    // ---- online softmax (row j+8*hi lives at acc index j over 16 lanes) ----
#pragma unroll
    for (int j = 0; j < 8; ++j) {
      float mx = -1e30f;
#pragma unroll
      for (int ni = 0; ni < 4; ++ni) {
        lg[ni][j] *= scale;
        mx = fmaxf(mx, lg[ni][j]);
      }
#pragma unroll
      for (int off = 8; off >= 1; off >>= 1)
        mx = fmaxf(mx, __shfl_xor(mx, off, 32));
      const float newm = fmaxf(mrow[j], mx);
      float s = 0.f;
#pragma unroll
      for (int ni = 0; ni < 4; ++ni) {
        const float p = __expf(lg[ni][j] - newm);
        lg[ni][j] = p;
        s += p;
      }
#pragma unroll
      for (int off = 8; off >= 1; off >>= 1) s += __shfl_xor(s, off, 32);
      const float alpha = __expf(mrow[j] - newm);
      lrow[j] = lrow[j] * alpha + s;
      mrow[j] = newm;
#pragma unroll
      for (int di = 0; di < 4; ++di) {
        o_r[di][j] *= alpha;
        o_i[di][j] *= alpha;
      }
    }
    // ---- stage P (bf16) in per-wave LDS, reload as A fragments ----
#pragma unroll
    for (int ni = 0; ni < 4; ++ni)
#pragma unroll
      for (int j = 0; j < 8; ++j)
        myP[(j + 8 * hi) * 64 + 16 * ni + lr] = (__bf16)lg[ni][j];

    // ---- O += P * V ----
#pragma unroll
    for (int t = 0; t < 2; ++t) {
      const v16bf fa = frag_a_bf(myP + lr * 64 + t * 32 + hi * 16);
#pragma unroll
      for (int di = 0; di < 4; ++di) {
        const size_t voff =
            base + (size_t)(n0 + t * 32 + hi * 16) * D_ + 16 * di + lr;
        o_r[di] = wmma_bf16(fa, frag_b_bf_strided(vr + voff, D_), o_r[di]);
        o_i[di] = wmma_bf16(fa, frag_b_bf_strided(vi + voff, D_), o_i[di]);
      }
    }
  }

#pragma unroll
  for (int j = 0; j < 8; ++j) {
    const float inv = 1.0f / lrow[j];
    const int row = q0 + j + 8 * hi;
#pragma unroll
    for (int di = 0; di < 4; ++di) {
      const size_t off = base + (size_t)row * D_ + 16 * di + lr;
      Or[off] = (__bf16)(o_r[di][j] * inv);
      Oi[off] = (__bf16)(o_i[di][j] * inv);
    }
  }
}

// ---------------------------------------------------------------------------
// Residual add + complex LayerNorm; emits fp32 (residual path) + bf16 (GEMM
// operand path). One 256-thread WG per (b,s) row. In-place safe on R.
// ---------------------------------------------------------------------------
__global__ __launch_bounds__(256) void cln_kernel(
    const float* __restrict__ Xr, const float* __restrict__ Xi,
    const float* __restrict__ Rr, const float* __restrict__ Ri,
    const float* __restrict__ grr, const float* __restrict__ gri,
    const float* __restrict__ gii, const float* __restrict__ bb_r,
    const float* __restrict__ bb_i, float* __restrict__ outR,
    float* __restrict__ outI, __bf16* __restrict__ outRb,
    __bf16* __restrict__ outIb) {
  const size_t row = blockIdx.x;
  const int t = threadIdx.x;
  const size_t i0 = row * D_ + t;
  const size_t i1 = i0 + 256;

  const float a0 = Xr[i0] + Rr[i0], a1 = Xr[i1] + Rr[i1];
  const float c0 = Xi[i0] + Ri[i0], c1 = Xi[i1] + Ri[i1];

  __shared__ float sm[5][256];
  sm[0][t] = a0 + a1;
  sm[1][t] = c0 + c1;
  sm[2][t] = a0 * a0 + a1 * a1;
  sm[3][t] = c0 * c0 + c1 * c1;
  sm[4][t] = a0 * c0 + a1 * c1;
  __syncthreads();
  for (int s = 128; s > 0; s >>= 1) {
    if (t < s)
#pragma unroll
      for (int q = 0; q < 5; ++q) sm[q][t] += sm[q][t + s];
    __syncthreads();
  }
  const float invD = 1.0f / (float)D_;
  const float mr = sm[0][0] * invD;
  const float mi = sm[1][0] * invD;
  const float Vrr = sm[2][0] * invD - mr * mr + 1e-5f;
  const float Vii = sm[3][0] * invD - mi * mi + 1e-5f;
  const float Vri = sm[4][0] * invD - mr * mi;
  const float s = sqrtf(Vrr * Vii - Vri * Vri);
  const float tt = sqrtf(Vrr + Vii + 2.0f * s);
  const float ist = 1.0f / (s * tt);
  const float Wrr = (Vii + s) * ist;
  const float Wii = (Vrr + s) * ist;
  const float Wri = -Vri * ist;

  const float cr0 = a0 - mr, ci0 = c0 - mi;
  const float cr1 = a1 - mr, ci1 = c1 - mi;
  const float hr0 = Wrr * cr0 + Wri * ci0, hi0 = Wri * cr0 + Wii * ci0;
  const float hr1 = Wrr * cr1 + Wri * ci1, hi1 = Wri * cr1 + Wii * ci1;
  const int g0 = t, g1 = t + 256;
  const float yr0 = grr[g0] * hr0 + gri[g0] * hi0 + bb_r[g0];
  const float yi0 = gri[g0] * hr0 + gii[g0] * hi0 + bb_i[g0];
  const float yr1 = grr[g1] * hr1 + gri[g1] * hi1 + bb_r[g1];
  const float yi1 = gri[g1] * hr1 + gii[g1] * hi1 + bb_i[g1];
  outR[i0] = yr0;
  outI[i0] = yi0;
  outR[i1] = yr1;
  outI[i1] = yi1;
  outRb[i0] = (__bf16)yr0;
  outIb[i0] = (__bf16)yi0;
  outRb[i1] = (__bf16)yr1;
  outIb[i1] = (__bf16)yi1;
}

// ---------------------------------------------------------------------------
// Orchestration
// ---------------------------------------------------------------------------
extern "C" void kernel_launch(void* const* d_in, const int* in_sizes, int n_in,
                              void* d_out, int out_size, void* d_ws,
                              size_t ws_size, hipStream_t stream) {
  (void)in_sizes;
  (void)n_in;
  (void)out_size;
  (void)ws_size;
  const float* xr_in = (const float*)d_in[0];
  const float* xi_in = (const float*)d_in[1];
  const float* bq_r = (const float*)d_in[14];
  const float* bq_i = (const float*)d_in[15];
  const float* bk_r = (const float*)d_in[16];
  const float* bk_i = (const float*)d_in[17];
  const float* bv_r = (const float*)d_in[18];
  const float* bv_i = (const float*)d_in[19];
  const float* bo_r = (const float*)d_in[20];
  const float* bo_i = (const float*)d_in[21];
  const float* b1_r = (const float*)d_in[22];
  const float* b1_i = (const float*)d_in[23];
  const float* b2_r = (const float*)d_in[24];
  const float* b2_i = (const float*)d_in[25];
  const float* ln1_grr = (const float*)d_in[26];
  const float* ln1_gii = (const float*)d_in[27];
  const float* ln2_grr = (const float*)d_in[28];
  const float* ln2_gii = (const float*)d_in[29];
  const float* ln1_gri = (const float*)d_in[30];
  const float* ln1_br = (const float*)d_in[31];
  const float* ln1_bi = (const float*)d_in[32];
  const float* ln2_gri = (const float*)d_in[33];
  const float* ln2_br = (const float*)d_in[34];
  const float* ln2_bi = (const float*)d_in[35];

  const size_t NE = (size_t)M_ * D_;          // activation elements
  const size_t NW = (size_t)L_ * D_ * D_;     // per-weight-tensor elements

  char* p = (char*)d_ws;
  float* Xr = (float*)p;  p += NE * 4;
  float* Xi = (float*)p;  p += NE * 4;
  float* Gr = (float*)p;  p += NE * 4;
  float* Gi = (float*)p;  p += NE * 4;
  __bf16* Xbr = (__bf16*)p; p += NE * 2;
  __bf16* Xbi = (__bf16*)p; p += NE * 2;
  __bf16* Qbr = (__bf16*)p; p += NE * 2;
  __bf16* Qbi = (__bf16*)p; p += NE * 2;
  __bf16* Kbr = (__bf16*)p; p += NE * 2;
  __bf16* Kbi = (__bf16*)p; p += NE * 2;
  __bf16* Vbr = (__bf16*)p; p += NE * 2;
  __bf16* Vbi = (__bf16*)p; p += NE * 2;
  __bf16* Tbr = (__bf16*)p; p += NE * 2;
  __bf16* Tbi = (__bf16*)p; p += NE * 2;
  __bf16* WB[12];
  for (int i = 0; i < 12; ++i) { WB[i] = (__bf16*)p; p += NW * 2; }

  // One-shot weight conversion fp32 -> bf16 (stays L2-resident afterwards).
  for (int i = 0; i < 12; ++i)
    cvt_kernel<<<dim3((unsigned)(NW / 2048)), dim3(256), 0, stream>>>(
        (const float*)d_in[2 + i], WB[i]);
  init_x_kernel<<<dim3((unsigned)(NE / 256)), dim3(256), 0, stream>>>(
      xr_in, xi_in, Xr, Xi, Xbr, Xbi);

  const dim3 gGrid(M_ / 64, D_ / 128), gBlk(256);
  const dim3 aGrid(B_ * H_ * (S_ / 64)), aBlk(128);
  const dim3 nGrid(M_), nBlk(256);

  for (int l = 0; l < L_; ++l) {
    const size_t wo = (size_t)l * D_ * D_;
    const size_t bo = (size_t)l * D_;
    // Q, K, V projections (bf16 outputs only)
    clin_kernel<<<gGrid, gBlk, 0, stream>>>(Xbr, Xbi, WB[0] + wo, WB[1] + wo,
                                            bq_r + bo, bq_i + bo, Gr, Gi, Qbr,
                                            Qbi, 2);
    clin_kernel<<<gGrid, gBlk, 0, stream>>>(Xbr, Xbi, WB[2] + wo, WB[3] + wo,
                                            bk_r + bo, bk_i + bo, Gr, Gi, Kbr,
                                            Kbi, 2);
    clin_kernel<<<gGrid, gBlk, 0, stream>>>(Xbr, Xbi, WB[4] + wo, WB[5] + wo,
                                            bv_r + bo, bv_i + bo, Gr, Gi, Vbr,
                                            Vbi, 2);
    // attention -> T (bf16)
    attn_kernel<<<aGrid, aBlk, 0, stream>>>(Qbr, Qbi, Kbr, Kbi, Vbr, Vbi, Tbr,
                                            Tbi);
    // output projection -> G (fp32 only)
    clin_kernel<<<gGrid, gBlk, 0, stream>>>(Tbr, Tbi, WB[6] + wo, WB[7] + wo,
                                            bo_r + bo, bo_i + bo, Gr, Gi, Qbr,
                                            Qbi, 1);
    // residual + LN1 -> X (fp32) + Xb (bf16)
    cln_kernel<<<nGrid, nBlk, 0, stream>>>(Gr, Gi, Xr, Xi, ln1_grr + bo,
                                           ln1_gri + bo, ln1_gii + bo,
                                           ln1_br + bo, ln1_bi + bo, Xr, Xi,
                                           Xbr, Xbi);
    // FFN: W1 + CReLU -> Q (bf16) ; W2 -> G (fp32) ; residual + LN2 -> X/Xb
    clin_kernel<<<gGrid, gBlk, 0, stream>>>(Xbr, Xbi, WB[8] + wo, WB[9] + wo,
                                            b1_r + bo, b1_i + bo, Gr, Gi, Qbr,
                                            Qbi, 2 | 4);
    clin_kernel<<<gGrid, gBlk, 0, stream>>>(Qbr, Qbi, WB[10] + wo, WB[11] + wo,
                                            b2_r + bo, b2_i + bo, Gr, Gi, Kbr,
                                            Kbi, 1);
    cln_kernel<<<nGrid, nBlk, 0, stream>>>(Gr, Gi, Xr, Xi, ln2_grr + bo,
                                           ln2_gri + bo, ln2_gii + bo,
                                           ln2_br + bo, ln2_bi + bo, Xr, Xi,
                                           Xbr, Xbi);
  }

  float* out = (float*)d_out;
  (void)hipMemcpyAsync(out, Xr, NE * sizeof(float), hipMemcpyDeviceToDevice,
                       stream);
  (void)hipMemcpyAsync(out + NE, Xi, NE * sizeof(float),
                       hipMemcpyDeviceToDevice, stream);
}
